// sccnn_conv_no_e_to_n_sc_1_8976481648972
// MI455X (gfx1250) — matrix operand order repaired and checked by reading, not compiled, not run on hardware
//
#include <hip/hip_runtime.h>

// ---------------------------------------------------------------------------
// SCCNN conv (node/edge Chebyshev streams) for gfx1250.
// Heavy GEMMs: fp32 operands streamed from HBM, split on the fly into
// bf16 (hi,lo) pairs, 3x v_wmma_f32_16x16x32_bf16 per tile (bf16x2 scheme)
// with f32 accumulation -> near-fp32 accuracy, memory-bound at ~23.3 TB/s.
// Block = 16 output rows, 8 waves split the K dimension (32-wide slabs of a
// 256-wide staged tile) -> 768 blocks on the 12288-row passes for occupancy.
// ---------------------------------------------------------------------------

typedef __attribute__((ext_vector_type(16))) __bf16 v16bf;
typedef __attribute__((ext_vector_type(8)))  __bf16 v8bf;
typedef __attribute__((ext_vector_type(4)))  __bf16 v4bf;
typedef __attribute__((ext_vector_type(8)))  float  v8f;

__device__ __forceinline__ v16bf mk16(v8bf lo, v8bf hi) {
    return __builtin_shufflevector(lo, hi, 0,1,2,3,4,5,6,7,8,9,10,11,12,13,14,15);
}

__device__ __forceinline__ v8f wmma_bf16(v16bf a, v16bf b, v8f c) {
    // (neg_a, A, neg_b, B, c_mod, C, reuse_a, reuse_b)
    return __builtin_amdgcn_wmma_f32_16x16x32_bf16(false, a, false, b, (short)0, c, false, false);
}

// split fp32 into bf16 hi + bf16 lo (bf16x2 decomposition)
__device__ __forceinline__ void split2(float v, __bf16& hi, __bf16& lo) {
    __bf16 h = (__bf16)v;
    float r = v - (float)h;   // exact: widening is exact
    hi = h;
    lo = (__bf16)r;
}

__device__ __forceinline__ void split4(const float* v, v4bf& h, v4bf& l) {
    #pragma unroll
    for (int j = 0; j < 4; ++j) {
        __bf16 hh, ll;
        split2(v[j], hh, ll);
        h[j] = hh;
        l[j] = ll;
    }
}

#define KT   256            // staged K extent per iteration
#define LDSS (KT + 8)       // padded LDS row stride in bf16 elems (528B, 16B mult)

// ---------------------------------------------------------------------------
// Y[row, 0:W] = alpha * (L @ X)[row, 0:W] + beta * Z[row, 0:W]
// L: n x n fp32 row-major. One block = 16 output rows; 8 waves each own a
// 32-wide K slab of the staged 16 x 256 A-tile; cross-wave sum via LDS.
// ---------------------------------------------------------------------------
template<int W>
__global__ __launch_bounds__(256)
void cheb_gemm_kernel(const float* __restrict__ L, int n,
                      const float* __restrict__ X, int ldX,
                      const float* __restrict__ Z, int ldZ,
                      float alpha, float beta,
                      float* __restrict__ Y, int ldY)
{
    static_assert(W == 16 || W == 32, "W must be 16 or 32");
    constexpr int NT   = W / 16;
    constexpr int AB   = 16 * LDSS * 2;     // bytes per A array
    constexpr int BB   = W * LDSS * 2;      // bytes per B array
    constexpr int NBG  = (W == 32) ? 8 : 4; // B k4-groups per thread

    __shared__ __align__(16) char smem[2 * AB + 2 * BB];
    __bf16* sAhi = (__bf16*)(smem);
    __bf16* sAlo = (__bf16*)(smem + AB);
    __bf16* sBhi = (__bf16*)(smem + 2 * AB);
    __bf16* sBlo = (__bf16*)(smem + 2 * AB + BB);
    float*  sRed = (float*)(smem);          // reused after the K loop (16KB max)

    const int tid  = threadIdx.x;
    const int wave = tid >> 5;
    const int lane = tid & 31;
    const int half = lane >> 4;
    const int mn   = lane & 15;
    const int row0 = blockIdx.x * 16;

    // B staging mapping: fixed column, gather 4 consecutive k (packed b64 store)
    const int cB  = tid & (W - 1);
    const int kgB = tid / W;                // base k4-group
    // A staging mapping: float4 along k
    // e = q*256+tid -> m = e>>6 (0..15), k4 = (e&63)*4

    v8f acc[NT];
    #pragma unroll
    for (int t = 0; t < NT; ++t) {
        v8f z = {0.f, 0.f, 0.f, 0.f, 0.f, 0.f, 0.f, 0.f};
        acc[t] = z;
    }

    for (int kbase = 0; kbase < n; kbase += KT) {
        // ---------------- load phase (batch all global loads) ----------------
        float4 va[4];
        #pragma unroll
        for (int q = 0; q < 4; ++q) {
            int e  = q * 256 + tid;
            int m  = e >> 6;
            int k4 = (e & 63) * 4;
            va[q] = *(const float4*)(L + (size_t)(row0 + m) * (size_t)n + kbase + k4);
        }
        float vb[NBG][4];
        #pragma unroll
        for (int q = 0; q < NBG; ++q) {
            int k4 = (kgB + q * (256 / W)) * 4;
            #pragma unroll
            for (int j = 0; j < 4; ++j)
                vb[q][j] = X[(size_t)(kbase + k4 + j) * (size_t)ldX + cB];
        }

        // ---------------- convert + packed LDS stores ----------------
        #pragma unroll
        for (int q = 0; q < 4; ++q) {
            int e  = q * 256 + tid;
            int m  = e >> 6;
            int k4 = (e & 63) * 4;
            float vv[4] = {va[q].x, va[q].y, va[q].z, va[q].w};
            v4bf h, l;
            split4(vv, h, l);
            *(v4bf*)(sAhi + m * LDSS + k4) = h;
            *(v4bf*)(sAlo + m * LDSS + k4) = l;
        }
        #pragma unroll
        for (int q = 0; q < NBG; ++q) {
            int k4 = (kgB + q * (256 / W)) * 4;
            v4bf h, l;
            split4(vb[q], h, l);
            *(v4bf*)(sBhi + cB * LDSS + k4) = h;
            *(v4bf*)(sBlo + cB * LDSS + k4) = l;
        }
        __syncthreads();

        // ---------------- WMMA over this wave's 32-wide K slab ----------------
        // A frag: lane row M=lane%16; VGPR v: K = 16*(v/4) + 8*half + 2*(v%4)
        const int ko  = wave * 32;
        const int aof = mn * LDSS + ko + half * 8;
        v16bf ahi = mk16(*(const v8bf*)(sAhi + aof), *(const v8bf*)(sAhi + aof + 16));
        v16bf alo = mk16(*(const v8bf*)(sAlo + aof), *(const v8bf*)(sAlo + aof + 16));

        // B frag: lane col N=lane%16; VGPR v: K = 16*(v/4) + 8*half + 2*(v%4)
        #pragma unroll
        for (int t = 0; t < NT; ++t) {
            const int bof = (t * 16 + mn) * LDSS + ko + half * 8;
            v16bf bhi = mk16(*(const v8bf*)(sBhi + bof), *(const v8bf*)(sBhi + bof + 16));
            v16bf blo = mk16(*(const v8bf*)(sBlo + bof), *(const v8bf*)(sBlo + bof + 16));
            acc[t] = wmma_bf16(ahi, bhi, acc[t]);   // hi*hi
            acc[t] = wmma_bf16(ahi, blo, acc[t]);   // hi*lo
            acc[t] = wmma_bf16(alo, bhi, acc[t]);   // lo*hi
        }
        __syncthreads();
    }

    // ---------------- cross-wave reduction via LDS ----------------
    // C layout: col N = lane%16, row M = r + 8*half
    #pragma unroll
    for (int t = 0; t < NT; ++t) {
        #pragma unroll
        for (int r = 0; r < 8; ++r) {
            int rowm = r + 8 * half;
            int col  = t * 16 + mn;
            sRed[(wave * 16 + rowm) * W + col] = acc[t][r];
        }
    }
    __syncthreads();

    #pragma unroll
    for (int e = tid; e < 16 * W; e += 256) {
        int rowm = e / W;
        int col  = e % W;
        float s = 0.f;
        #pragma unroll
        for (int w = 0; w < 8; ++w)
            s += sRed[(w * 16 + rowm) * W + col];
        int row = row0 + rowm;
        float v = alpha * s;
        if (Z) v += beta * Z[(size_t)row * (size_t)ldZ + col];
        Y[(size_t)row * (size_t)ldY + col] = v;
    }
}

// ---------------------------------------------------------------------------
// x1n[j, f] = D2diag[j] * sum_i B1[i, j] * x0[i, f] / D1diag[i]
// B1: n0 x n1 row-major. Output rows are n1 (j); K dim is n0 (i).
// Same 16-row / wave-K-split structure as cheb_gemm (W = 16).
// ---------------------------------------------------------------------------
__global__ __launch_bounds__(256)
void proj_gemm_kernel(const float* __restrict__ B1,
                      const float* __restrict__ x0,
                      const float* __restrict__ D1,  // n0 x n0 (diag used)
                      const float* __restrict__ D2,  // n1 x n1 (diag used)
                      float* __restrict__ x1n,
                      int n0, int n1)
{
    constexpr int AB = 16 * LDSS * 2;
    constexpr int BB = 16 * LDSS * 2;

    __shared__ __align__(16) char smem[2 * AB + 2 * BB];
    __bf16* sAhi = (__bf16*)(smem);
    __bf16* sAlo = (__bf16*)(smem + AB);
    __bf16* sBhi = (__bf16*)(smem + 2 * AB);
    __bf16* sBlo = (__bf16*)(smem + 2 * AB + BB);
    float*  sRed = (float*)(smem);

    const int tid  = threadIdx.x;
    const int wave = tid >> 5;
    const int lane = tid & 31;
    const int half = lane >> 4;
    const int mn   = lane & 15;
    const int j0   = blockIdx.x * 16;

    // Both A and B staged by gathering 4 consecutive k at a fixed column:
    // A column = output row m (A[m][k] = B1[k, j0+m]); B column = feature f.
    const int cS  = tid & 15;
    const int kgS = tid >> 4;   // 0..15

    v8f acc = {0.f, 0.f, 0.f, 0.f, 0.f, 0.f, 0.f, 0.f};

    for (int kbase = 0; kbase < n0; kbase += KT) {
        float vaA[4][4], vaB[4][4];
        #pragma unroll
        for (int q = 0; q < 4; ++q) {
            int k4 = (kgS + q * 16) * 4;
            #pragma unroll
            for (int j = 0; j < 4; ++j) {
                size_t k = (size_t)(kbase + k4 + j);
                vaA[q][j] = B1[k * (size_t)n1 + j0 + cS];
                vaB[q][j] = x0[k * 16 + cS] / D1[k * (size_t)n0 + k];
            }
        }
        #pragma unroll
        for (int q = 0; q < 4; ++q) {
            int k4 = (kgS + q * 16) * 4;
            v4bf h, l;
            split4(vaA[q], h, l);
            *(v4bf*)(sAhi + cS * LDSS + k4) = h;
            *(v4bf*)(sAlo + cS * LDSS + k4) = l;
            split4(vaB[q], h, l);
            *(v4bf*)(sBhi + cS * LDSS + k4) = h;
            *(v4bf*)(sBlo + cS * LDSS + k4) = l;
        }
        __syncthreads();

        const int ko  = wave * 32;
        const int aof = mn * LDSS + ko + half * 8;
        v16bf ahi = mk16(*(const v8bf*)(sAhi + aof), *(const v8bf*)(sAhi + aof + 16));
        v16bf alo = mk16(*(const v8bf*)(sAlo + aof), *(const v8bf*)(sAlo + aof + 16));

        const int bof = mn * LDSS + ko + half * 8;
        v16bf bhi = mk16(*(const v8bf*)(sBhi + bof), *(const v8bf*)(sBhi + bof + 16));
        v16bf blo = mk16(*(const v8bf*)(sBlo + bof), *(const v8bf*)(sBlo + bof + 16));

        acc = wmma_bf16(ahi, bhi, acc);
        acc = wmma_bf16(ahi, blo, acc);
        acc = wmma_bf16(alo, bhi, acc);
        __syncthreads();
    }

    #pragma unroll
    for (int r = 0; r < 8; ++r) {
        int rowm = r + 8 * half;
        sRed[(wave * 16 + rowm) * 16 + mn] = acc[r];
    }
    __syncthreads();

    {
        int rowm = tid >> 4;   // 256 threads cover 16x16 exactly
        int col  = tid & 15;
        float s = 0.f;
        #pragma unroll
        for (int w = 0; w < 8; ++w)
            s += sRed[(w * 16 + rowm) * 16 + col];
        int j = j0 + rowm;
        x1n[j * 16 + col] = s * D2[(size_t)j * (size_t)n1 + j];
    }
}

// ---------------------------------------------------------------------------
// s0[j, 0:32] = [x1n[j, 0:16] | x1[j, 0:16]]
// ---------------------------------------------------------------------------
__global__ void pack_s0_kernel(const float* __restrict__ x1n,
                               const float* __restrict__ x1,
                               float* __restrict__ s0, int n1)
{
    int idx = blockIdx.x * 256 + threadIdx.x;
    if (idx >= n1 * 32) return;
    int j = idx >> 5;
    int c = idx & 31;
    s0[idx] = (c < 16) ? x1n[j * 16 + c] : x1[j * 16 + (c - 16)];
}

// ---------------------------------------------------------------------------
// y0[n,o] = relu( sum_{k=0..3} sum_i tap_k[n,i] * W0[i,o,k] )
// ---------------------------------------------------------------------------
__global__ void out0_kernel(const float* __restrict__ x0,
                            const float* __restrict__ t1,
                            const float* __restrict__ t2,
                            const float* __restrict__ t3,
                            const float* __restrict__ W0,
                            float* __restrict__ y0)
{
    int idx  = blockIdx.x * 256 + threadIdx.x;  // 4096*16
    int nrow = idx >> 4;
    int o    = idx & 15;
    float acc = 0.f;
    #pragma unroll
    for (int i = 0; i < 16; ++i) {
        const float* w = W0 + (i * 16 + o) * 4;
        acc += x0[nrow * 16 + i] * w[0];
        acc += t1[nrow * 16 + i] * w[1];
        acc += t2[nrow * 16 + i] * w[2];
        acc += t3[nrow * 16 + i] * w[3];
    }
    y0[idx] = fmaxf(acc, 0.f);
}

// ---------------------------------------------------------------------------
// y1[n,o] = relu( sum_{k=0..6} sum_i tap_k[n,i] * W1[i,o,k] )
// taps: x1n, s1[:,0:16], s2[:,0:16], x1, s1[:,16:32], s2[:,16:32], s3
// ---------------------------------------------------------------------------
__global__ void out1_kernel(const float* __restrict__ x1n,
                            const float* __restrict__ s1,
                            const float* __restrict__ s2,
                            const float* __restrict__ x1,
                            const float* __restrict__ s3,
                            const float* __restrict__ W1,
                            float* __restrict__ y1)
{
    int idx  = blockIdx.x * 256 + threadIdx.x;  // 12288*16
    int nrow = idx >> 4;
    int o    = idx & 15;
    float acc = 0.f;
    #pragma unroll
    for (int i = 0; i < 16; ++i) {
        const float* w = W1 + (i * 16 + o) * 7;
        acc += x1n[nrow * 16 + i]      * w[0];
        acc += s1[nrow * 32 + i]       * w[1];
        acc += s2[nrow * 32 + i]       * w[2];
        acc += x1[nrow * 16 + i]       * w[3];
        acc += s1[nrow * 32 + 16 + i]  * w[4];
        acc += s2[nrow * 32 + 16 + i]  * w[5];
        acc += s3[nrow * 16 + i]       * w[6];
    }
    y1[idx] = fmaxf(acc, 0.f);
}

// ---------------------------------------------------------------------------

extern "C" void kernel_launch(void* const* d_in, const int* in_sizes, int n_in,
                              void* d_out, int out_size, void* d_ws, size_t ws_size,
                              hipStream_t stream)
{
    const int N0 = 4096, N1 = 12288;

    const float* x0  = (const float*)d_in[0];   // [4096,16]
    const float* x1  = (const float*)d_in[1];   // [12288,16]
    const float* B1  = (const float*)d_in[2];   // [4096,12288]
    const float* L0  = (const float*)d_in[3];   // [4096,4096]
    const float* L1l = (const float*)d_in[4];   // [12288,12288]
    const float* D1  = (const float*)d_in[5];   // [4096,4096] diag
    const float* D2  = (const float*)d_in[6];   // [12288,12288] diag
    const float* W0  = (const float*)d_in[7];   // [16,16,4]
    const float* W1  = (const float*)d_in[8];   // [16,16,7]

    float* out = (float*)d_out;                 // y0 (65536) then y1 (196608)

    float* ws    = (float*)d_ws;
    float* t1_0  = ws;                // [4096,16]
    float* t2_0  = ws + 65536;        // [4096,16]
    float* t3_0  = ws + 131072;       // [4096,16]
    float* x1n   = ws + 196608;       // [12288,16]
    float* s0    = ws + 393216;       // [12288,32]
    float* s1    = ws + 786432;       // [12288,32]
    float* s2    = ws + 1179648;      // [12288,32]
    float* s3    = ws + 1572864;      // [12288,16]

    // x1n = D2 * (B1^T * (D1^-1 * x0))
    proj_gemm_kernel<<<N1 / 16, 256, 0, stream>>>(B1, x0, D1, D2, x1n, N0, N1);

    // node-stream Chebyshev taps on L0
    cheb_gemm_kernel<16><<<N0 / 16, 256, 0, stream>>>(L0, N0, x0,   16, nullptr, 0, 1.f,  0.f, t1_0, 16);
    cheb_gemm_kernel<16><<<N0 / 16, 256, 0, stream>>>(L0, N0, t1_0, 16, x0,     16, 2.f, -1.f, t2_0, 16);
    cheb_gemm_kernel<16><<<N0 / 16, 256, 0, stream>>>(L0, N0, t2_0, 16, t1_0,   16, 2.f, -1.f, t3_0, 16);

    // fuse the two edge streams into 32 columns so L1l is streamed only 3x
    pack_s0_kernel<<<(N1 * 32) / 256, 256, 0, stream>>>(x1n, x1, s0, N1);
    cheb_gemm_kernel<32><<<N1 / 16, 256, 0, stream>>>(L1l, N1, s0, 32, nullptr, 0, 1.f,  0.f, s1, 32);
    cheb_gemm_kernel<32><<<N1 / 16, 256, 0, stream>>>(L1l, N1, s1, 32, s0,     32, 2.f, -1.f, s2, 32);
    // third tap only for the x1 stream (columns 16:32)
    cheb_gemm_kernel<16><<<N1 / 16, 256, 0, stream>>>(L1l, N1, s2 + 16, 32, s1 + 16, 32, 2.f, -1.f, s3, 16);

    // weight contractions + ReLU
    out0_kernel<<<(N0 * 16) / 256, 256, 0, stream>>>(x0, t1_0, t2_0, t3_0, W0, out);
    out1_kernel<<<(N1 * 16) / 256, 256, 0, stream>>>(x1n, s1, s2, x1, s3, W1, out + N0 * 16);
}
